// GCN_classifier_39591008534852
// MI455X (gfx1250) — compile-verified, matrix operand order checked by reading, not run on hardware
//
#include <hip/hip_runtime.h>
#include <hip/hip_bf16.h>

typedef __attribute__((ext_vector_type(16))) __bf16 v16bf;
typedef __attribute__((ext_vector_type(8)))  __bf16 v8bf;
typedef __attribute__((ext_vector_type(8)))  float  v8f;

#define B_BATCH 256
#define N_NODES 512
#define F_IN    3
#define H_DIM   64
#define C_CLS   10

// D = A(16x32 bf16) x B(32x16 bf16) + C(f32), wave32 CDNA5 WMMA
__device__ __forceinline__ v8f wmma_bf16(v16bf a, v16bf b, v8f c) {
    return __builtin_amdgcn_wmma_f32_16x16x32_bf16(
        /*neg_a=*/false, a, /*neg_b=*/false, b,
        /*c_mod=*/(short)0, c, /*reuse_a=*/false, /*reuse_b=*/false);
}

// Preload this lane's 16 K-positions for one K-tile (shared by all 4 M-tiles).
// pk[i] corresponds to node k = kbase + (i&7) + ((i&8)<<1), kbase = kt*32 + khalf*8.
__device__ __forceinline__ void load_ktile_pos(const float2* s_pos, int kbase,
                                               float2 pk[16]) {
    const float4* p0 = (const float4*)(s_pos + kbase);        // k = kbase+0..7
    const float4* p1 = (const float4*)(s_pos + kbase + 16);   // k = kbase+16..23
#pragma unroll
    for (int q = 0; q < 4; ++q) {
        const float4 a = p0[q];
        pk[2 * q]     = make_float2(a.x, a.y);
        pk[2 * q + 1] = make_float2(a.z, a.w);
        const float4 b = p1[q];
        pk[8 + 2 * q]     = make_float2(b.x, b.y);
        pk[8 + 2 * q + 1] = make_float2(b.z, b.w);
    }
}

// Straight-line adjacency A-tile build (16-bit A VGPR layout, wave32).
// Raw v_rsq_f32 (no denorm wrapper) + v_cndmask for the diagonal: no branches.
__device__ __forceinline__ v16bf make_adj_tile(float2 pm, int m, int kbase,
                                               const float2 pk[16]) {
    v16bf a;
#pragma unroll
    for (int i = 0; i < 16; ++i) {
        const int k = kbase + (i & 7) + ((i & 8) << 1);
        const float dx = pm.x - pk[i].x;
        const float dy = pm.y - pk[i].y;
        const float d2 = dx * dx + dy * dy;
        float v = __builtin_amdgcn_rsqf(d2) * (1.0f / 29.0f);
        v = (k == m) ? 1.0f : v;   // diagonal -> cndmask
        a[i] = (__bf16)v;
    }
    return a;
}

// ---------------------------------------------------------------------------
// Kernel 1: XW1 = x @ W1, bf16 pre-swizzled in the per-lane WMMA B-layout:
// xw1[b][kt(16)][ht(4)][lane(32)][16 bf16]. One wave per (b,kt,ht) tile.
// ---------------------------------------------------------------------------
__global__ void __launch_bounds__(128) k_xw1(const float* __restrict__ x,
                                             const float* __restrict__ W1,
                                             __bf16* __restrict__ xw1) {
    const int tid  = threadIdx.x;
    const int lane = tid & 31;
    const int wv   = tid >> 5;
    const int tile = blockIdx.x * 4 + wv;   // 0 .. B*64-1
    const int b  = tile >> 6;
    const int kt = (tile >> 2) & 15;
    const int ht = tile & 3;

    const int h     = ht * 16 + (lane & 15);
    const int khalf = (lane >> 4) & 1;
    const float w0 = W1[0 * H_DIM + h];
    const float w1 = W1[1 * H_DIM + h];
    const float w2 = W1[2 * H_DIM + h];

    v16bf bt;
#pragma unroll
    for (int i = 0; i < 16; ++i) {
        const int m = kt * 32 + khalf * 8 + (i & 7) + ((i & 8) << 1);
        const float* xr = x + ((size_t)b * N_NODES + m) * F_IN;
        const float v = xr[0] * w0 + xr[1] * w1 + xr[2] * w2;
        bt[i] = (__bf16)v;
    }
    *(v16bf*)(xw1 + ((((size_t)b * 16 + kt) * 4 + ht) * 32 + lane) * 16) = bt;
}

// ---------------------------------------------------------------------------
// Kernel 2: h = relu(adj @ XW1 + b1), then HW2 = h @ W2 emitted bf16 in
// pass-2's swizzled B-layout hw2[b][kt(16)][lane(32)][16] (classes padded).
// Block = 128 threads (4 waves) = half a batch's rows; wave: 4 Mt x 4 Ht tiles.
// ---------------------------------------------------------------------------
__global__ void __launch_bounds__(128, 1) k_pass1(const float* __restrict__ x,
                                                  const __bf16* __restrict__ xw1,
                                                  const float* __restrict__ b1,
                                                  const float* __restrict__ W2,
                                                  __bf16* __restrict__ hw2) {
    __shared__ __align__(16) float2 s_pos[N_NODES];
    __shared__ __align__(16) __bf16 s_h[4][64][64];   // per-wave h slab (bf16)

    const int tid   = threadIdx.x;
    const int lane  = tid & 31;
    const int wv    = tid >> 5;
    const int b     = blockIdx.x >> 1;
    const int mhalf = blockIdx.x & 1;

    for (int n = tid; n < N_NODES; n += 128) {
        const float* xr = x + ((size_t)b * N_NODES + n) * F_IN;
        s_pos[n] = make_float2(xr[0], xr[1]);
    }
    __syncthreads();

    const int mt0   = mhalf * 16 + wv * 4;
    const int mloc  = lane & 15;
    const int khalf = (lane >> 4) & 1;
    float2 pm[4];
#pragma unroll
    for (int j = 0; j < 4; ++j) pm[j] = s_pos[(mt0 + j) * 16 + mloc];

    v8f acc[4][4] = {};
    const __bf16* btb = xw1 + (size_t)b * 16 * 4 * 32 * 16;

    for (int kt = 0; kt < 16; ++kt) {
        v16bf bt[4];
#pragma unroll
        for (int ht = 0; ht < 4; ++ht)
            bt[ht] = *(const v16bf*)(btb + (((size_t)kt * 4 + ht) * 32 + lane) * 16);
        if (kt < 15)  // near-scope prefetch of next K-slab
            __builtin_prefetch(btb + (((size_t)(kt + 1) * 4) * 32 + lane) * 16, 0, 3);

        const int kbase = kt * 32 + khalf * 8;
        float2 pk[16];
        load_ktile_pos(s_pos, kbase, pk);   // shared by all 4 M-tiles
#pragma unroll
        for (int j = 0; j < 4; ++j) {
            const v16bf a = make_adj_tile(pm[j], (mt0 + j) * 16 + mloc, kbase, pk);
#pragma unroll
            for (int ht = 0; ht < 4; ++ht)
                acc[j][ht] = wmma_bf16(a, bt[ht], acc[j][ht]);
        }
    }

    // bias + relu -> per-wave LDS slab (D-layout: VGPR r, lane L -> row r+8*(L>>4), col L&15)
#pragma unroll
    for (int j = 0; j < 4; ++j)
#pragma unroll
        for (int ht = 0; ht < 4; ++ht) {
            const int col = ht * 16 + (lane & 15);
            const float bias = b1[col];
#pragma unroll
            for (int r = 0; r < 8; ++r) {
                const int rowl = j * 16 + r + ((lane >> 4) << 3);
                s_h[wv][rowl][col] = (__bf16)fmaxf(acc[j][ht][r] + bias, 0.0f);
            }
        }
    __syncthreads();

    // HW2 = h @ W2, written directly in pass-2's swizzled B-layout.
    const int c = lane & 15;
    float w2c[64];
#pragma unroll
    for (int hh = 0; hh < 64; ++hh)
        w2c[hh] = (c < C_CLS) ? W2[hh * C_CLS + c] : 0.0f;

#pragma unroll
    for (int t = 0; t < 2; ++t) {
        const int ktg = mhalf * 8 + wv * 2 + t;   // global K-tile these rows feed
        v16bf bt;
#pragma unroll
        for (int i = 0; i < 16; ++i) {
            const int lr = t * 32 + khalf * 8 + (i & 7) + ((i & 8) << 1);
            const v8bf* rowp = (const v8bf*)&s_h[wv][lr][0];
            float s = 0.0f;
#pragma unroll
            for (int ch = 0; ch < 8; ++ch) {
                const v8bf q = rowp[ch];
#pragma unroll
                for (int e = 0; e < 8; ++e)
                    s += (float)q[e] * w2c[ch * 8 + e];
            }
            bt[i] = (__bf16)s;
        }
        *(v16bf*)(hw2 + (((size_t)b * 16 + ktg) * 32 + lane) * 16) = bt;
    }
}

// ---------------------------------------------------------------------------
// Kernel 3: out_pre = adj @ HW2 (f32, B x 512 x 16). b2 is constant along the
// softmax axis (axis=1) and cancels exactly, so it is skipped.
// ---------------------------------------------------------------------------
__global__ void __launch_bounds__(128, 1) k_pass2(const float* __restrict__ x,
                                                  const __bf16* __restrict__ hw2,
                                                  float* __restrict__ out_pre) {
    __shared__ __align__(16) float2 s_pos[N_NODES];
    const int tid   = threadIdx.x;
    const int lane  = tid & 31;
    const int wv    = tid >> 5;
    const int b     = blockIdx.x >> 1;
    const int mhalf = blockIdx.x & 1;

    for (int n = tid; n < N_NODES; n += 128) {
        const float* xr = x + ((size_t)b * N_NODES + n) * F_IN;
        s_pos[n] = make_float2(xr[0], xr[1]);
    }
    __syncthreads();

    const int mt0   = mhalf * 16 + wv * 4;
    const int mloc  = lane & 15;
    const int khalf = (lane >> 4) & 1;
    float2 pm[4];
#pragma unroll
    for (int j = 0; j < 4; ++j) pm[j] = s_pos[(mt0 + j) * 16 + mloc];

    v8f acc[4] = {};
    for (int kt = 0; kt < 16; ++kt) {
        const v16bf bt = *(const v16bf*)(hw2 + (((size_t)b * 16 + kt) * 32 + lane) * 16);
        const int kbase = kt * 32 + khalf * 8;
        float2 pk[16];
        load_ktile_pos(s_pos, kbase, pk);
#pragma unroll
        for (int j = 0; j < 4; ++j) {
            const v16bf a = make_adj_tile(pm[j], (mt0 + j) * 16 + mloc, kbase, pk);
            acc[j] = wmma_bf16(a, bt, acc[j]);
        }
    }
    const int col = lane & 15;
#pragma unroll
    for (int j = 0; j < 4; ++j)
#pragma unroll
        for (int r = 0; r < 8; ++r) {
            const int row = (mt0 + j) * 16 + r + ((lane >> 4) << 3);
            out_pre[((size_t)b * N_NODES + row) * 16 + col] = acc[j][r];
        }
}

// ---------------------------------------------------------------------------
// Kernel 4: log_softmax over axis=1 (the 512 nodes) per (batch, class).
// One block (256 threads) per batch; two-stage LDS reduction.
// ---------------------------------------------------------------------------
__global__ void __launch_bounds__(256) k_logsoftmax(const float* __restrict__ pre,
                                                    float* __restrict__ out) {
    __shared__ float red[16][16];
    __shared__ float s_max[16];
    __shared__ float s_log[16];
    const int b = blockIdx.x;
    const int t = threadIdx.x;
    const int c = t & 15;
    const int g = t >> 4;
    const float* pb = pre + (size_t)b * N_NODES * 16;

    float m = -1e30f;
    for (int i = 0; i < 32; ++i) m = fmaxf(m, pb[(g + i * 16) * 16 + c]);
    red[g][c] = m;
    __syncthreads();
    if (t < 16) {
        float mm = -1e30f;
        for (int gg = 0; gg < 16; ++gg) mm = fmaxf(mm, red[gg][t]);
        s_max[t] = mm;
    }
    __syncthreads();
    const float cm = s_max[c];
    float s = 0.0f;
    for (int i = 0; i < 32; ++i) s += expf(pb[(g + i * 16) * 16 + c] - cm);
    red[g][c] = s;
    __syncthreads();
    if (t < 16) {
        float ss = 0.0f;
        for (int gg = 0; gg < 16; ++gg) ss += red[gg][t];
        s_log[t] = logf(ss);
    }
    __syncthreads();
    for (int j = 0; j < 2; ++j) {
        const int n = t * 2 + j;
        const float* row = pb + (size_t)n * 16;
        float* orow = out + ((size_t)b * N_NODES + n) * C_CLS;
#pragma unroll
        for (int cc = 0; cc < C_CLS; ++cc)
            orow[cc] = row[cc] - s_max[cc] - s_log[cc];
    }
}

extern "C" void kernel_launch(void* const* d_in, const int* in_sizes, int n_in,
                              void* d_out, int out_size, void* d_ws, size_t ws_size,
                              hipStream_t stream) {
    (void)in_sizes; (void)n_in; (void)out_size; (void)ws_size;
    const float* x  = (const float*)d_in[0];
    const float* W1 = (const float*)d_in[1];
    const float* b1 = (const float*)d_in[2];
    const float* W2 = (const float*)d_in[3];
    // d_in[4] (b2) cancels in axis-1 log_softmax -> unused.
    float* out = (float*)d_out;

    // Workspace layout (28 MB total, L2-resident on MI455X's 192 MB L2):
    //   [ 0,16MB)  xw1 swizzled bf16 : B*16*4*32*16 = 8,388,608 elems
    //   [16,20MB)  hw2 swizzled bf16 : B*16*32*16   = 2,097,152 elems
    //   [20,28MB)  pre-softmax f32   : B*512*16     = 2,097,152 elems
    char* ws = (char*)d_ws;
    __bf16* xw1 = (__bf16*)ws;
    __bf16* hw2 = (__bf16*)(ws + (size_t)16 * 1024 * 1024);
    float*  pre = (float*) (ws + (size_t)20 * 1024 * 1024);

    k_xw1       <<<dim3(B_BATCH * 64 / 4), dim3(128), 0, stream>>>(x, W1, xw1);
    k_pass1     <<<dim3(B_BATCH * 2),      dim3(128), 0, stream>>>(x, xw1, b1, W2, hw2);
    k_pass2     <<<dim3(B_BATCH * 2),      dim3(128), 0, stream>>>(x, hw2, pre);
    k_logsoftmax<<<dim3(B_BATCH),          dim3(256), 0, stream>>>(pre, out);
}
